// MoEEP_54546084659886
// MI455X (gfx1250) — matrix operand order repaired and checked by reading, not compiled, not run on hardware
//
#include <hip/hip_runtime.h>
#include <hip/hip_bf16.h>
#include <math.h>

// ---------------- types ----------------
typedef _Float16 f16;
typedef __attribute__((ext_vector_type(16))) _Float16 v16h;
typedef __attribute__((ext_vector_type(8)))  _Float16 v8h;
typedef __attribute__((ext_vector_type(8)))  float    v8f;

#define N_TOK   16384
#define IN_D    1024
#define HID_D   2048
#define N_EXP   16
#define CAP     8192   // capacity_factor 1.0 * N / top_k

__device__ __forceinline__ v16h frag_cat(v8h lo, v8h hi) {
  return __builtin_shufflevector(lo, hi, 0,1,2,3,4,5,6,7,8,9,10,11,12,13,14,15);
}
__device__ __forceinline__ v8f wmma16x16x32(v16h a, v16h b, v8f c) {
  // (neg_a, A, neg_b, B, c_mod, C, reuse_a, reuse_b) -> v_wmma_f32_16x16x32_f16
  return __builtin_amdgcn_wmma_f32_16x16x32_f16(false, a, false, b, (short)0, c, false, false);
}
__device__ __forceinline__ float gelu_erf(float v) {
  return 0.5f * v * (1.0f + erff(v * 0.70710678118654752f));
}

// ---------------- init ----------------
__global__ void zero_f4_kernel(float* __restrict__ p, long n4) {
  long i = (long)blockIdx.x * blockDim.x + threadIdx.x;
  if (i < n4) ((float4*)p)[i] = make_float4(0.f, 0.f, 0.f, 0.f);
}
__global__ void zero_small_kernel(float* probs_sum, float* counts_raw) {
  int i = threadIdx.x;
  if (i < N_EXP) { probs_sum[i] = 0.f; counts_raw[i] = 0.f; }
}

// ------- weight pre-transpose + f32->f16 convert: in[K][N] -> out[N][K] -----
__global__ __launch_bounds__(256) void transpose_f16_kernel(
    const float* __restrict__ in, f16* __restrict__ out, int K, int N) {
  __shared__ f16 tile[32][33];
  const int tx = threadIdx.x & 31, ty = threadIdx.x >> 5;  // ty in [0,8)
  const int n0 = blockIdx.x * 32, k0 = blockIdx.y * 32;
#pragma unroll
  for (int j = 0; j < 4; ++j) {
    const int k = k0 + ty + j * 8;
    tile[ty + j * 8][tx] = (f16)in[(long)k * N + n0 + tx];  // coalesced read
  }
  __syncthreads();
#pragma unroll
  for (int j = 0; j < 4; ++j) {
    const int n = n0 + ty + j * 8;
    out[(long)n * K + k0 + tx] = tile[tx][ty + j * 8];      // coalesced write
  }
}

// ---------------- router: logits -> softmax -> top2 + aux accumulators ------
__global__ __launch_bounds__(256) void router_kernel(
    const float* __restrict__ x, const float* __restrict__ rw,
    const float* __restrict__ rb,
    int* __restrict__ topk_idx, float* __restrict__ topk_w,
    float* __restrict__ probs_sum, float* __restrict__ counts_raw) {
  const int wave = threadIdx.x >> 5, lane = threadIdx.x & 31;
  const int t = blockIdx.x * 8 + wave;
  float acc[N_EXP];
#pragma unroll
  for (int e = 0; e < N_EXP; ++e) acc[e] = 0.f;
  const float* xr = x + (long)t * IN_D;
  for (int i = lane; i < IN_D; i += 32) {
    float xv = xr[i];
    const float* wr = rw + i * N_EXP;
#pragma unroll
    for (int e = 0; e < N_EXP; ++e) acc[e] += xv * wr[e];
  }
#pragma unroll
  for (int e = 0; e < N_EXP; ++e)
#pragma unroll
    for (int off = 16; off >= 1; off >>= 1)
      acc[e] += __shfl_xor(acc[e], off, 32);
  if (lane == 0) {
    float p[N_EXP];
    float mx = -1e30f;
#pragma unroll
    for (int e = 0; e < N_EXP; ++e) { p[e] = acc[e] + rb[e]; mx = fmaxf(mx, p[e]); }
    float s = 0.f;
#pragma unroll
    for (int e = 0; e < N_EXP; ++e) { p[e] = __expf(p[e] - mx); s += p[e]; }
    float inv = 1.f / s;
#pragma unroll
    for (int e = 0; e < N_EXP; ++e) p[e] *= inv;
    // top-2, lowest-index tie-break (matches lax.top_k)
    int i1 = 0;
#pragma unroll
    for (int e = 1; e < N_EXP; ++e) if (p[e] > p[i1]) i1 = e;
    int i2 = (i1 == 0) ? 1 : 0;
#pragma unroll
    for (int e = 0; e < N_EXP; ++e) if (e != i1 && p[e] > p[i2]) i2 = e;
    topk_idx[t * 2 + 0] = i1; topk_idx[t * 2 + 1] = i2;
    topk_w[t * 2 + 0] = p[i1]; topk_w[t * 2 + 1] = p[i2];
#pragma unroll
    for (int e = 0; e < N_EXP; ++e) atomicAdd(&probs_sum[e], p[e]);
    atomicAdd(&counts_raw[i1], 1.f);
    atomicAdd(&counts_raw[i2], 1.f);
  }
}

// -------- dispatch: token-order ballot scan, one wave per expert ------------
__global__ __launch_bounds__(512) void dispatch_kernel(
    const int* __restrict__ topk_idx, const float* __restrict__ topk_w,
    int* __restrict__ slot_token, float* __restrict__ slot_w,
    int* __restrict__ counts_cap) {
  const int e = threadIdx.x >> 5, lane = threadIdx.x & 31;
  int base = 0;
  for (int t0 = 0; t0 < N_TOK; t0 += 32) {
    const int t = t0 + lane;
    const int a = topk_idx[t * 2 + 0], b = topk_idx[t * 2 + 1];
    const bool m = (a == e) || (b == e);
    const unsigned b32 = (unsigned)__ballot((int)m);
    const int prefix = __popc(b32 & ((1u << lane) - 1u));
    if (m) {
      const int pos = base + prefix;
      if (pos < CAP) {
        slot_token[e * CAP + pos] = t;
        slot_w[e * CAP + pos] = (a == e) ? topk_w[t * 2] : topk_w[t * 2 + 1];
      }
    }
    base += __popc(b32);
  }
  if (lane == 0) counts_cap[e] = (base < CAP) ? base : CAP;
}

// ---------------- shared GEMM fragment helpers ------------------------------
// A LDS tile: [128][32] f16 row-major. B LDS tile: [128][32] f16, n-major
// (pre-transposed weights). Fragment layouts per ISA 7.12.2.
__device__ __forceinline__ v16h load_a_frag(const f16* As, int m, int half) {
  v8h lo = *(const v8h*)(As + m * 32 + 8 * half);
  v8h hi = *(const v8h*)(As + m * 32 + 16 + 8 * half);
  return frag_cat(lo, hi);
}
__device__ __forceinline__ v16h load_b_frag(const f16* Bs, int n, int half) {
  v8h lo = *(const v8h*)(Bs + n * 32 + 8 * half);
  v8h hi = *(const v8h*)(Bs + n * 32 + 16 + 8 * half);
  return frag_cat(lo, hi);
}

// -------- GEMM1: H = gelu(gather(x)[CAP,1024] @ w1[e] + b1[e]) --------------
// block tile 128(M) x 128(N), 8 waves (4Mx2N), wave tile 32x64 = 2x4 wmma
__global__ __launch_bounds__(256) void gemm1_kernel(
    const float* __restrict__ x, const f16* __restrict__ Bt,  // [HID_D][IN_D]
    const float* __restrict__ b1, const int* __restrict__ slot_token,
    const int* __restrict__ counts_cap, f16* __restrict__ Hbuf, int e) {
  __shared__ alignas(16) f16 As[128 * 32];
  __shared__ alignas(16) f16 Bs[128 * 32];
  const int count = counts_cap[e];
  const int bm = blockIdx.x;
  if (bm * 128 >= count) return;
  const int n0 = blockIdx.y * 128;
  const float* b1e = b1 + e * HID_D;
  const int* st = slot_token + e * CAP;

  const int tid = threadIdx.x, lane = tid & 31, wave = tid >> 5;
  const int wm = (wave & 3) * 32, wn = (wave >> 2) * 64;
  const int half = lane >> 4, l16 = lane & 15;

  // staging assignments: each thread moves 16 f16 for A and 16 f16 for B
  const int arow = tid >> 1, akh = (tid & 1) * 16;
  const int s_row = bm * 128 + arow;
  int tok = -1;
  if (s_row < count) tok = st[s_row];
  const int brow = tid >> 1, bkh = (tid & 1) * 16;
  const f16* bsrc_base = Bt + (long)(n0 + brow) * IN_D + bkh;

  v8f acc[2][4] = {{{}, {}, {}, {}}, {{}, {}, {}, {}}};

  for (int k0 = 0; k0 < IN_D; k0 += 32) {
    __syncthreads();
    {  // stage A tile, f32 -> f16
      v8h lo, hi;
      if (tok >= 0) {
        const float4* xp = (const float4*)(x + (long)tok * IN_D + k0 + akh);
        float4 v0 = xp[0], v1 = xp[1], v2 = xp[2], v3 = xp[3];
        lo[0]=(f16)v0.x; lo[1]=(f16)v0.y; lo[2]=(f16)v0.z; lo[3]=(f16)v0.w;
        lo[4]=(f16)v1.x; lo[5]=(f16)v1.y; lo[6]=(f16)v1.z; lo[7]=(f16)v1.w;
        hi[0]=(f16)v2.x; hi[1]=(f16)v2.y; hi[2]=(f16)v2.z; hi[3]=(f16)v2.w;
        hi[4]=(f16)v3.x; hi[5]=(f16)v3.y; hi[6]=(f16)v3.z; hi[7]=(f16)v3.w;
      } else {
#pragma unroll
        for (int j = 0; j < 8; ++j) { lo[j] = (f16)0.f; hi[j] = (f16)0.f; }
      }
      v8h* dst = (v8h*)(As + arow * 32 + akh);
      dst[0] = lo; dst[1] = hi;
    }
    {  // stage B tile: wide f16 copy from pre-transposed weights
      const v8h* src = (const v8h*)(bsrc_base + k0);
      v8h* dst = (v8h*)(Bs + brow * 32 + bkh);
      dst[0] = src[0]; dst[1] = src[1];
      if (k0 + 32 < IN_D) __builtin_prefetch(bsrc_base + k0 + 32, 0, 0);
    }
    __syncthreads();
    v16h afrag[2], bfrag[4];
#pragma unroll
    for (int tm = 0; tm < 2; ++tm) afrag[tm] = load_a_frag(As, wm + tm * 16 + l16, half);
#pragma unroll
    for (int tn = 0; tn < 4; ++tn) bfrag[tn] = load_b_frag(Bs, wn + tn * 16 + l16, half);
#pragma unroll
    for (int tm = 0; tm < 2; ++tm)
#pragma unroll
      for (int tn = 0; tn < 4; ++tn)
        acc[tm][tn] = wmma16x16x32(afrag[tm], bfrag[tn], acc[tm][tn]);
  }
  // epilogue: bias + exact GELU -> H (f16)
#pragma unroll
  for (int tm = 0; tm < 2; ++tm)
#pragma unroll
    for (int tn = 0; tn < 4; ++tn)
#pragma unroll
      for (int r = 0; r < 8; ++r) {
        const int m = wm + tm * 16 + 8 * half + r;
        const int s = bm * 128 + m;
        if (s < count) {
          const int n = n0 + wn + tn * 16 + l16;
          float v = acc[tm][tn][r] + b1e[n];
          Hbuf[(long)s * HID_D + n] = (f16)gelu_erf(v);
        }
      }
}

// -------- GEMM2: y[token] += w_tok * (H @ w2[e] + b2[e])  (fused scatter) ---
__global__ __launch_bounds__(256) void gemm2_kernel(
    const f16* __restrict__ Hbuf, const f16* __restrict__ Bt,  // [IN_D][HID_D]
    const float* __restrict__ b2, const int* __restrict__ slot_token,
    const float* __restrict__ slot_w, const int* __restrict__ counts_cap,
    float* __restrict__ y, int e) {
  __shared__ alignas(16) f16 As[128 * 32];
  __shared__ alignas(16) f16 Bs[128 * 32];
  const int count = counts_cap[e];
  const int bm = blockIdx.x;
  if (bm * 128 >= count) return;
  const int n0 = blockIdx.y * 128;
  const float* b2e = b2 + e * IN_D;
  const int* st = slot_token + e * CAP;
  const float* sw = slot_w + e * CAP;

  const int tid = threadIdx.x, lane = tid & 31, wave = tid >> 5;
  const int wm = (wave & 3) * 32, wn = (wave >> 2) * 64;
  const int half = lane >> 4, l16 = lane & 15;

  const int arow = tid >> 1, akh = (tid & 1) * 16;
  const int s_row = bm * 128 + arow;   // < CAP always
  const f16* asrc_base = Hbuf + (long)s_row * HID_D + akh;
  const int brow = tid >> 1, bkh = (tid & 1) * 16;
  const f16* bsrc_base = Bt + (long)(n0 + brow) * HID_D + bkh;

  v8f acc[2][4] = {{{}, {}, {}, {}}, {{}, {}, {}, {}}};

  for (int k0 = 0; k0 < HID_D; k0 += 32) {
    __syncthreads();
    {  // stage A from H (f16), wide copy
      const v8h* src = (const v8h*)(asrc_base + k0);
      v8h* dst = (v8h*)(As + arow * 32 + akh);
      dst[0] = src[0]; dst[1] = src[1];
    }
    {  // stage B from pre-transposed weights, wide copy
      const v8h* src = (const v8h*)(bsrc_base + k0);
      v8h* dst = (v8h*)(Bs + brow * 32 + bkh);
      dst[0] = src[0]; dst[1] = src[1];
      if (k0 + 32 < HID_D) __builtin_prefetch(bsrc_base + k0 + 32, 0, 0);
    }
    __syncthreads();
    v16h afrag[2], bfrag[4];
#pragma unroll
    for (int tm = 0; tm < 2; ++tm) afrag[tm] = load_a_frag(As, wm + tm * 16 + l16, half);
#pragma unroll
    for (int tn = 0; tn < 4; ++tn) bfrag[tn] = load_b_frag(Bs, wn + tn * 16 + l16, half);
#pragma unroll
    for (int tm = 0; tm < 2; ++tm)
#pragma unroll
      for (int tn = 0; tn < 4; ++tn)
        acc[tm][tn] = wmma16x16x32(afrag[tm], bfrag[tn], acc[tm][tn]);
  }
  // epilogue: weighted scatter-add into y (slots within an expert are unique
  // tokens; expert kernels serialize on the stream -> plain RMW is safe)
#pragma unroll
  for (int tm = 0; tm < 2; ++tm)
#pragma unroll
    for (int tn = 0; tn < 4; ++tn)
#pragma unroll
      for (int r = 0; r < 8; ++r) {
        const int m = wm + tm * 16 + 8 * half + r;
        const int s = bm * 128 + m;
        if (s < count) {
          const int n = n0 + wn + tn * 16 + l16;
          const int t = st[s];
          const float wt = sw[s];
          float v = acc[tm][tn][r] + b2e[n];
          float* yp = y + (long)t * IN_D + n;
          *yp = *yp + wt * v;
        }
      }
}

// ---------------- aux loss ----------------
__global__ void aux_kernel(const float* probs_sum, const float* counts_raw,
                           float* out_aux) {
  const int lane = threadIdx.x;
  float bal = 0.f, imp = 0.f;
  if (lane < N_EXP) {
    float p = probs_sum[lane], c = counts_raw[lane];
    bal = (p / (float)N_TOK) * (c / (float)N_TOK);  // density * usage
    imp = p * p;                                    // importance^2
  }
  for (int off = 16; off >= 1; off >>= 1) {
    bal += __shfl_xor(bal, off, 32);
    imp += __shfl_xor(imp, off, 32);
  }
  if (lane == 0) *out_aux = bal * (float)N_EXP + imp / (float)N_EXP;
}

// ---------------- host launcher ----------------
extern "C" void kernel_launch(void* const* d_in, const int* in_sizes, int n_in,
                              void* d_out, int out_size, void* d_ws,
                              size_t ws_size, hipStream_t stream) {
  const float* x  = (const float*)d_in[0];
  const float* rw = (const float*)d_in[1];
  const float* rb = (const float*)d_in[2];
  const float* w1 = (const float*)d_in[3];
  const float* b1 = (const float*)d_in[4];
  const float* w2 = (const float*)d_in[5];
  const float* b2 = (const float*)d_in[6];
  float* y = (float*)d_out;                       // [N_TOK*IN_D] + aux scalar

  char* ws = (char*)d_ws;
  int*   topk_idx   = (int*)(ws);                          // 128 KB
  float* topk_w     = (float*)(ws + (128 << 10));          // 128 KB
  float* probs_sum  = (float*)(ws + (256 << 10));          // 64 B
  float* counts_raw = (float*)(ws + (256 << 10) + 64);     // 64 B
  int*   counts_cap = (int*)(ws + (256 << 10) + 128);      // 64 B
  int*   slot_token = (int*)(ws + (1 << 20));              // 512 KB
  float* slot_w     = (float*)(ws + (1 << 20) + (512 << 10));
  f16*   Hbuf       = (f16*)(ws + (2l << 20));             // 32 MB
  f16*   W1t        = (f16*)(ws + (34l << 20));            // 4 MB  [HID][IN]
  f16*   W2t        = (f16*)(ws + (38l << 20));            // 4 MB  [IN][HID]

  // 1) zero y and small accumulators
  {
    long n4 = (long)N_TOK * IN_D / 4;
    zero_f4_kernel<<<(unsigned)((n4 + 255) / 256), 256, 0, stream>>>(y, n4);
    zero_small_kernel<<<1, 32, 0, stream>>>(probs_sum, counts_raw);
  }
  // 2) router
  router_kernel<<<N_TOK / 8, 256, 0, stream>>>(x, rw, rb, topk_idx, topk_w,
                                               probs_sum, counts_raw);
  // 3) dispatch (token-order scan)
  dispatch_kernel<<<1, 512, 0, stream>>>(topk_idx, topk_w, slot_token, slot_w,
                                         counts_cap);
  // 4) expert FFNs (sequential on stream: fused scatter needs no atomics)
  for (int e = 0; e < N_EXP; ++e) {
    transpose_f16_kernel<<<dim3(HID_D / 32, IN_D / 32), 256, 0, stream>>>(
        w1 + (long)e * IN_D * HID_D, W1t, IN_D, HID_D);
    transpose_f16_kernel<<<dim3(IN_D / 32, HID_D / 32), 256, 0, stream>>>(
        w2 + (long)e * HID_D * IN_D, W2t, HID_D, IN_D);
    gemm1_kernel<<<dim3(CAP / 128, HID_D / 128), 256, 0, stream>>>(
        x, W1t, b1, slot_token, counts_cap, Hbuf, e);
    gemm2_kernel<<<dim3(CAP / 128, IN_D / 128), 256, 0, stream>>>(
        Hbuf, W2t, b2, slot_token, slot_w, counts_cap, y, e);
  }
  // 5) aux loss -> last output element
  aux_kernel<<<1, 32, 0, stream>>>(probs_sum, counts_raw,
                                   y + (long)N_TOK * IN_D);
}
